// TernaryLinear_22050362097970
// MI455X (gfx1250) — matrix operand order repaired and checked by reading, not compiled, not run on hardware
//
#include <hip/hip_runtime.h>
#include <hip/hip_bf16.h>
#include <stdint.h>

typedef __attribute__((ext_vector_type(16))) _Float16 v16h;
typedef __attribute__((ext_vector_type(8)))  _Float16 v8h;
typedef __attribute__((ext_vector_type(8)))  float    v8f;
typedef __attribute__((ext_vector_type(4)))  float    v4f;

#define D_IN   2048
#define D_OUT  2048
#define M_TOTAL 16384            // 8 * 2048 rows
#define WELEMS ((size_t)D_IN * D_OUT)
#define LN_EPS 1e-5f

// ---------------- Kernel 1a: partial sum of |W| (deterministic, no atomics) ----------------
__global__ __launch_bounds__(256) void k_abs_partial(const float* __restrict__ W,
                                                     float* __restrict__ partial) {
  __shared__ float red[256];
  const int tid = threadIdx.x;
  const size_t gid    = (size_t)blockIdx.x * blockDim.x + tid;
  const size_t stride = (size_t)gridDim.x * blockDim.x;
  float s = 0.f;
  for (size_t i = gid * 4; i < WELEMS; i += stride * 4) {
    v4f v = *(const v4f*)(W + i);
    s += fabsf(v[0]) + fabsf(v[1]) + fabsf(v[2]) + fabsf(v[3]);
  }
  red[tid] = s;
  __syncthreads();
  for (int off = 128; off > 0; off >>= 1) {
    if (tid < off) red[tid] += red[tid + off];
    __syncthreads();
  }
  if (tid == 0) partial[blockIdx.x] = red[0];
}

// ---------------- Kernel 1b: finalize -> inv_scale = numel / sum(|W|) ----------------
__global__ __launch_bounds__(256) void k_abs_final(const float* __restrict__ partial,
                                                   float* __restrict__ inv_scale) {
  __shared__ float red[256];
  const int tid = threadIdx.x;
  red[tid] = partial[tid];
  __syncthreads();
  for (int off = 128; off > 0; off >>= 1) {
    if (tid < off) red[tid] += red[tid + off];
    __syncthreads();
  }
  if (tid == 0) inv_scale[0] = (float)WELEMS / red[0];
}

// ---------------- Kernel 2: ternary quantize + transpose  W[K][N] -> Wqt[N][K] f16 ----------------
__global__ __launch_bounds__(256) void k_quant_transpose(const float* __restrict__ W,
                                                         const float* __restrict__ inv_scale_p,
                                                         _Float16* __restrict__ Wqt) {
  __shared__ float tile[32][33];
  const float inv = inv_scale_p[0];
  const int tx = threadIdx.x;            // 0..31
  const int ty = threadIdx.y;            // 0..7
  const int n = blockIdx.x * 32 + tx;    // column of W
  const int k = blockIdx.y * 32 + ty;    // row of W
  #pragma unroll
  for (int j = 0; j < 32; j += 8)
    tile[ty + j][tx] = W[(size_t)(k + j) * D_OUT + n];
  __syncthreads();
  const int k2 = blockIdx.y * 32 + tx;   // contiguous K on write
  const int n2 = blockIdx.x * 32 + ty;
  #pragma unroll
  for (int j = 0; j < 32; j += 8) {
    float v = tile[tx][ty + j] * inv;
    v = fminf(fmaxf(rintf(v), -1.f), 1.f);   // rint = round-half-even, matches jnp.round
    Wqt[(size_t)(n2 + j) * D_IN + k2] = (_Float16)v;
  }
}

// ---------------- Kernel 3: LayerNorm row-wise, f32 -> f16 ----------------
__global__ __launch_bounds__(256) void k_layernorm(const float* __restrict__ x,
                                                   const float* __restrict__ g,
                                                   const float* __restrict__ b,
                                                   _Float16* __restrict__ xn) {
  __shared__ float s1[256];
  __shared__ float s2[256];
  const int tid = threadIdx.x;
  const size_t row = blockIdx.x;
  const float* xr = x + row * (size_t)D_IN;
  const int base = tid * 8;

  v4f a = *(const v4f*)(xr + base);
  v4f c = *(const v4f*)(xr + base + 4);
  float vals[8] = {a[0], a[1], a[2], a[3], c[0], c[1], c[2], c[3]};

  float sum = 0.f, sq = 0.f;
  #pragma unroll
  for (int i = 0; i < 8; ++i) { sum += vals[i]; sq += vals[i] * vals[i]; }
  s1[tid] = sum; s2[tid] = sq;
  __syncthreads();
  for (int off = 128; off > 0; off >>= 1) {
    if (tid < off) { s1[tid] += s1[tid + off]; s2[tid] += s2[tid + off]; }
    __syncthreads();
  }
  const float mean = s1[0] * (1.f / D_IN);
  const float var  = s2[0] * (1.f / D_IN) - mean * mean;   // biased variance
  const float rstd = rsqrtf(var + LN_EPS);

  v4f g0 = *(const v4f*)(g + base);
  v4f g1 = *(const v4f*)(g + base + 4);
  v4f b0 = *(const v4f*)(b + base);
  v4f b1 = *(const v4f*)(b + base + 4);
  float gs[8] = {g0[0], g0[1], g0[2], g0[3], g1[0], g1[1], g1[2], g1[3]};
  float bs[8] = {b0[0], b0[1], b0[2], b0[3], b1[0], b1[1], b1[2], b1[3]};

  v8h o;
  #pragma unroll
  for (int i = 0; i < 8; ++i)
    o[i] = (_Float16)((vals[i] - mean) * rstd * gs[i] + bs[i]);
  *(v8h*)(xn + row * (size_t)D_IN + base) = o;
}

// ---------------- Kernel 4: WMMA GEMM  out = xn[M][K] @ Wqt[N][K]^T ----------------
// Block tile 256(M) x 128(N), BK=32, 8 waves of 64x64, double-buffered LDS filled by
// CDNA5 async global->LDS loads (ASYNCcnt).
#define BM 256
#define BN 128
#define BK 32
#define LDA 40                   // padded LDS row stride in halves (conflict avoidance)
#define ABUF (BM * LDA)          // halves per A buffer
#define BBUF (BN * LDA)          // halves per B buffer

__device__ __forceinline__ void async_ld_b128(uint32_t lds_byte_off, const void* gptr) {
  // per-lane: LDS[lds_byte_off] <= MEM[gptr] (16 bytes), tracked by ASYNCcnt
  asm volatile("global_load_async_to_lds_b128 %0, %1, off"
               :: "v"(lds_byte_off), "v"((uint64_t)(uintptr_t)gptr)
               : "memory");
}
__device__ __forceinline__ void wait_async0() {
  asm volatile("s_wait_asynccnt 0x0" ::: "memory");
}

__global__ __launch_bounds__(256, 1) void k_wmma_gemm(const _Float16* __restrict__ A,
                                                      const _Float16* __restrict__ Bt,
                                                      float* __restrict__ C) {
  __shared__ _Float16 sA[2 * ABUF];   // 40 KB
  __shared__ _Float16 sB[2 * BBUF];   // 20 KB

  const int tid  = threadIdx.x;
  const int lane = tid & 31;
  const int w    = tid >> 5;          // wave id 0..7
  const int wm   = w & 3;             // 64-row slab
  const int wn   = w >> 2;            // 64-col slab
  const int block_m = blockIdx.y * BM;
  const int block_n = blockIdx.x * BN;

  // staging chunks: thread -> (row = id>>2, col = (id&3)*8), 4 chunks A + 2 chunks B
  const int r0 = tid >> 2;            // 0..63
  const int c0 = (tid & 3) * 8;       // 0,8,16,24

  const uint32_t sAo = (uint32_t)(uintptr_t)&sA[0];   // LDS byte offset of sA
  const uint32_t sBo = (uint32_t)(uintptr_t)&sB[0];

  const _Float16* gA[4];
  uint32_t lA[4];
  #pragma unroll
  for (int p = 0; p < 4; ++p) {
    gA[p] = A + (size_t)(block_m + r0 + 64 * p) * D_IN + c0;
    lA[p] = sAo + (uint32_t)(((r0 + 64 * p) * LDA + c0) * 2);
  }
  const _Float16* gB[2];
  uint32_t lB[2];
  #pragma unroll
  for (int p = 0; p < 2; ++p) {
    gB[p] = Bt + (size_t)(block_n + r0 + 64 * p) * D_IN + c0;
    lB[p] = sBo + (uint32_t)(((r0 + 64 * p) * LDA + c0) * 2);
  }

  v8f acc[4][4];
  v8f zero = {};
  #pragma unroll
  for (int i = 0; i < 4; ++i)
    #pragma unroll
    for (int j = 0; j < 4; ++j) acc[i][j] = zero;

  // prime the pipeline: slab 0 -> buffer 0
  #pragma unroll
  for (int p = 0; p < 4; ++p) async_ld_b128(lA[p], gA[p]);
  #pragma unroll
  for (int p = 0; p < 2; ++p) async_ld_b128(lB[p], gB[p]);

  // fragment addressing per CDNA5 wave32 VGPR layouts
  const int fm  = lane & 15;
  const int kba = (lane < 16) ? 0 : 8;    // A: halves 0-7 <-> K kba..+7, halves 8-15 <-> K kba+16..+23
  const int kbb = (lane < 16) ? 0 : 8;    // B: halves 0-7 <-> K kbb..+7, halves 8-15 <-> K kbb+16..+23

  for (int kt = 0; kt < D_IN; kt += BK) {
    const int buf = (kt >> 5) & 1;
    wait_async0();       // our async writes into `buf` have landed in LDS
    __syncthreads();     // everyone's writes landed; previous reads of buf^1 done

    if (kt + BK < D_IN) {
      const int nb = buf ^ 1;
      #pragma unroll
      for (int p = 0; p < 4; ++p)
        async_ld_b128(lA[p] + (uint32_t)(nb * (ABUF * 2)), gA[p] + kt + BK);
      #pragma unroll
      for (int p = 0; p < 2; ++p)
        async_ld_b128(lB[p] + (uint32_t)(nb * (BBUF * 2)), gB[p] + kt + BK);
    }

    const _Float16* bA = sA + buf * ABUF;
    const _Float16* bB = sB + buf * BBUF;

    v16h af[4];
    #pragma unroll
    for (int i = 0; i < 4; ++i) {
      const v8h lo = *(const v8h*)&bA[(wm * 64 + i * 16 + fm) * LDA + kba];
      const v8h hi = *(const v8h*)&bA[(wm * 64 + i * 16 + fm) * LDA + kba + 16];
      af[i] = __builtin_shufflevector(lo, hi, 0,1,2,3,4,5,6,7,8,9,10,11,12,13,14,15);
    }
    v16h bf[4];
    #pragma unroll
    for (int j = 0; j < 4; ++j) {
      const v8h lo = *(const v8h*)&bB[(wn * 64 + j * 16 + fm) * LDA + kbb];
      const v8h hi = *(const v8h*)&bB[(wn * 64 + j * 16 + fm) * LDA + kbb + 16];
      bf[j] = __builtin_shufflevector(lo, hi, 0,1,2,3,4,5,6,7,8,9,10,11,12,13,14,15);
    }

    #pragma unroll
    for (int i = 0; i < 4; ++i)
      #pragma unroll
      for (int j = 0; j < 4; ++j)
        acc[i][j] = __builtin_amdgcn_wmma_f32_16x16x32_f16(
            /*neg_a=*/false, af[i], /*neg_b=*/false, bf[j],
            /*c_mod=*/(short)0, acc[i][j], /*reuse_a=*/false, /*reuse_b=*/false);
  }

  // epilogue: C/D layout -> VGPR r: lanes 0-15 row M=r, lanes 16-31 row M=8+r; lane&15 = column
  const int mhi = (lane < 16) ? 0 : 8;
  const int nn  = lane & 15;
  #pragma unroll
  for (int i = 0; i < 4; ++i)
    #pragma unroll
    for (int j = 0; j < 4; ++j) {
      float* cp = C + (size_t)(block_m + wm * 64 + i * 16 + mhi) * D_OUT
                    + block_n + wn * 64 + j * 16 + nn;
      #pragma unroll
      for (int r = 0; r < 8; ++r)
        __builtin_nontemporal_store(acc[i][j][r], cp + (size_t)r * D_OUT);
    }
}

// ---------------- Host launcher ----------------
extern "C" void kernel_launch(void* const* d_in, const int* in_sizes, int n_in,
                              void* d_out, int out_size, void* d_ws, size_t ws_size,
                              hipStream_t stream) {
  const float* x    = (const float*)d_in[0];   // [8,2048,2048]
  const float* W    = (const float*)d_in[1];   // [2048,2048]
  const float* ln_w = (const float*)d_in[2];   // [2048]
  const float* ln_b = (const float*)d_in[3];   // [2048]
  float* out = (float*)d_out;                  // [8,2048,2048] f32

  char* ws = (char*)d_ws;
  float*     inv_scale = (float*)ws;                 // 1 float
  float*     partial   = (float*)(ws + 256);         // 256 floats
  _Float16*  xn        = (_Float16*)(ws + 4096);                                   // 64 MB
  _Float16*  wqt       = (_Float16*)(ws + 4096 + (size_t)M_TOTAL * D_IN * 2);      // 8 MB

  k_abs_partial<<<256, 256, 0, stream>>>(W, partial);
  k_abs_final<<<1, 256, 0, stream>>>(partial, inv_scale);
  k_quant_transpose<<<dim3(D_OUT / 32, D_IN / 32), dim3(32, 8), 0, stream>>>(W, inv_scale, wqt);
  k_layernorm<<<M_TOTAL, 256, 0, stream>>>(x, ln_w, ln_b, xn);
  k_wmma_gemm<<<dim3(D_OUT / BN, M_TOTAL / BM), 256, 0, stream>>>(xn, wqt, out);
}
